// AWD_CRF_55774445305982
// MI455X (gfx1250) — matrix operand order repaired and checked by reading, not compiled, not run on hardware
//
#include <hip/hip_runtime.h>
#include <hip/hip_bf16.h>
#include <stdint.h>

// ---------------------------------------------------------------------------
// BiLSTM-CRF forward for MI455X (gfx1250), wave32.
//  * Input / output projections: v_wmma_f32_16x16x32_bf16 (f32 accumulate)
//  * LSTM recurrence: persistent kernel, W_hh slice pulled into LDS with
//    GLOBAL_LOAD_ASYNC_TO_LDS_B128 (ASYNCcnt), cross-WG step barrier via
//    agent-scope atomics + s_sleep, global_prefetch for next-step xg.
//  * CRF: single small block, sequential logsumexp scan.
// ---------------------------------------------------------------------------

#define T_LEN 4096
#define E_DIM 400
#define EPAD  416          // 13 * 32 (WMMA K pad)
#define H2    576
#define HD    1152         // 36 * 32
#define G4    2304         // 4 * H2
#define KTAG  34
#define KPAD  48           // 3 * 16 (WMMA N pad)
#define START_TAG 32
#define STOP_TAG  33
#define NEGV  (-1e4f)

#define REC_WGS     18     // workgroups per direction; 18*32 = 576 cells
#define ROWS_PER_WG 128    // 4 gates x 32 cells, one row per thread
#define WSTRIDE     578    // LDS row stride in bf16 elems (odd dword stride)

// async-copy block: per-(dir,wg) LDS image of the W_hh slice, padded so the
// whole block is 73 iterations of 128 lanes x 16B.
#define WL_USH        (ROWS_PER_WG * WSTRIDE)          // 73984 ushorts
#define ABLK_BYTES    149504                           // 73 * 128 * 16
#define ABLK_USH      (ABLK_BYTES / 2)                 // 74752
#define ABLK_ITERS    73
#define LDS_HDR_BYTES 2816                             // hbuf(2304) + gates(512)

typedef __attribute__((ext_vector_type(16))) __bf16 v16bf;
typedef __attribute__((ext_vector_type(8)))  float  v8f;
typedef __attribute__((ext_vector_type(4)))  unsigned int u32x4;

union FragBF { v16bf v; u32x4 q[2]; };

__device__ __forceinline__ unsigned short f2bf(float x) {
    unsigned int u = __float_as_uint(x);
    unsigned int r = u + 0x7FFFu + ((u >> 16) & 1u);   // round-to-nearest-even
    return (unsigned short)(r >> 16);
}
__device__ __forceinline__ float sigmoidf(float x) { return 1.f / (1.f + __expf(-x)); }

// ---------------------------------------------------------------------------
// 0) init: zero cross-WG step counters
__global__ void k_init(unsigned int* ctr) {
    if (threadIdx.x < 8) ctr[threadIdx.x] = 0u;
}

// 1) embedding gather + f32->bf16, pad E 400->416 with zeros
__global__ void k_embed(const int* __restrict__ sent, const float* __restrict__ emb,
                        unsigned short* __restrict__ xbf) {
    const int t = blockIdx.x;
    const int idx = sent[t];
    const float* src = emb + (size_t)idx * E_DIM;
    unsigned short* dst = xbf + (size_t)t * EPAD;
    for (int e = threadIdx.x; e < EPAD; e += blockDim.x)
        dst[e] = (e < E_DIM) ? f2bf(src[e]) : (unsigned short)0;
}

// 2) stage W_ih (both dirs) to bf16 [2][G4][EPAD], K-padded
__global__ void k_stage_wih(const float* __restrict__ wf, const float* __restrict__ wb,
                            unsigned short* __restrict__ out) {
    const int r = blockIdx.x, dir = blockIdx.y;
    const float* src = (dir ? wb : wf) + (size_t)r * E_DIM;
    unsigned short* dst = out + ((size_t)dir * G4 + r) * EPAD;
    for (int e = threadIdx.x; e < EPAD; e += blockDim.x)
        dst[e] = (e < E_DIM) ? f2bf(src[e]) : (unsigned short)0;
}

// 3) bsum[dir][G4] = b_ih + b_hh (folded into GEMM accumulator init)
__global__ void k_stage_bias(const float* bif, const float* bhf,
                             const float* bib, const float* bhb, float* bsum) {
    const int i = blockIdx.x * blockDim.x + threadIdx.x;
    if (i < G4) {
        bsum[i]      = bif[i] + bhf[i];
        bsum[G4 + i] = bib[i] + bhb[i];
    }
}

// 4) stage W_h2t to bf16 [KPAD][HD], N-padded rows zero
__global__ void k_stage_w2(const float* __restrict__ w2, unsigned short* __restrict__ out) {
    const int r = blockIdx.x;
    unsigned short* dst = out + (size_t)r * HD;
    const float* src = w2 + (size_t)r * HD;
    for (int e = threadIdx.x; e < HD; e += blockDim.x)
        dst[e] = (r < KTAG) ? f2bf(src[e]) : (unsigned short)0;
}

// 5) stage W_hh as per-(dir,wg) LDS images: bf16, row stride 578, block padded
//    to ABLK_USH with zeros, ready for raw async byte copy into LDS.
__global__ void k_stage_whh(const float* __restrict__ whh_f, const float* __restrict__ whh_b,
                            unsigned short* __restrict__ out) {
    const int wg = blockIdx.x, dir = blockIdx.y;
    const float* whh = dir ? whh_b : whh_f;
    const int cellBase = wg * 32;
    unsigned short* dst = out + (size_t)(dir * REC_WGS + wg) * ABLK_USH;
    for (int idx = threadIdx.x; idx < ABLK_USH; idx += blockDim.x) {
        const int r = idx / WSTRIDE;          // row in slice (valid when idx < WL_USH)
        const int k = idx - r * WSTRIDE;
        unsigned short v = 0;
        if (idx < WL_USH && k < H2) {
            const int rg = (r >> 5) * H2 + cellBase + (r & 31);
            v = f2bf(whh[(size_t)rg * H2 + k]);
        }
        dst[idx] = v;
    }
}

// ---------------------------------------------------------------------------
// 6) Input projection: xg[dir][t][n] = x[t'] . Wih[dir][n] + bsum[dir][n]
//    dir==1 uses reversed time rows of x (x[::-1]).
//    One 16x16 output tile per wave; K loop of 13 * v_wmma_f32_16x16x32_bf16.
__global__ void __launch_bounds__(128)
k_gemm_in(const unsigned short* __restrict__ xbf, const unsigned short* __restrict__ wih,
          const float* __restrict__ bsum, float* __restrict__ xg) {
    const int dir  = blockIdx.z;
    const int lane = threadIdx.x & 31;
    const int wv   = threadIdx.x >> 5;
    const int l16  = lane & 15;
    const int hi   = lane >> 4;                 // half-wave select
    const int mBase = blockIdx.x * 16;
    const int nBase = (blockIdx.y * 4 + wv) * 16;

    int rowA = mBase + l16;
    if (dir) rowA = T_LEN - 1 - rowA;
    const int rowB = nBase + l16;               // N column this lane owns in C/D
    const unsigned short* W = wih + (size_t)dir * G4 * EPAD;
    const float bias = bsum[dir * G4 + rowB];

    v8f c;
    #pragma unroll
    for (int r = 0; r < 8; ++r) c[r] = bias;    // bias depends only on N -> same in all 8

    const unsigned short* aRow = xbf + (size_t)rowA * EPAD;
    const unsigned short* bRow = W   + (size_t)rowB * EPAD;

    for (int k0 = 0; k0 < EPAD; k0 += 32) {
        FragBF a, b;
        const unsigned short* ap = aRow + k0 + hi * 8;   // K runs {off..off+7},{off+16..}
        a.q[0] = *(const u32x4*)(ap);
        a.q[1] = *(const u32x4*)(ap + 16);
        const unsigned short* bp = bRow + k0 + hi * 8;   // K runs {off..off+7},{off+16..}
        b.q[0] = *(const u32x4*)(bp);
        b.q[1] = *(const u32x4*)(bp + 16);
        c = __builtin_amdgcn_wmma_f32_16x16x32_bf16(false, a.v, false, b.v,
                                                    (short)0, c, false, false);
    }
    float* out = xg + (size_t)dir * T_LEN * G4;
    const int mOff = mBase + hi * 8;
    #pragma unroll
    for (int r = 0; r < 8; ++r)
        out[(size_t)(mOff + r) * G4 + nBase + l16] = c[r];
}

// ---------------------------------------------------------------------------
// 7) Persistent bidirectional LSTM recurrence.
//    grid = (18, 2); block = 128 (4 waves). WG owns cells [32*wg, 32*wg+32),
//    thread t owns gate row (t>>5)*576 + cellBase + (t&31).
//    LDS: [hbuf 576 f32][gates 128 f32][W_hh slice 128x578 bf16], the slice
//    filled with GLOBAL_LOAD_ASYNC_TO_LDS_B128 from the pre-staged image.
//    h double-buffered in global (h_pub[s&1][dir]); per-step device barrier =
//    release fetch_add + acquire spin (s_sleep) per direction.
__global__ void __launch_bounds__(128, 1)
k_lstm_rec(const unsigned short* __restrict__ whhS,
           const float* __restrict__ xg, float* __restrict__ h_pub,
           unsigned short* __restrict__ hcat, unsigned int* __restrict__ ctr) {
    extern __shared__ char smem[];
    float* hbuf  = (float*)smem;                                   // 576 f32
    float* gates = hbuf + H2;                                      // 128 f32
    unsigned short* wl = (unsigned short*)(smem + LDS_HDR_BYTES);  // 128*578 bf16 (+pad)

    const int tid = threadIdx.x;
    const int wg  = blockIdx.x;
    const int dir = blockIdx.y;
    const int cellBase = wg * 32;
    const int cl = tid & 31;
    const int row = (tid >> 5) * H2 + cellBase + cl;

    // --- async DMA of this WG's W_hh LDS image (149504 B, 73 x 128 x 16B) ---
    {
        const char* gsrc = (const char*)(whhS + (size_t)(dir * REC_WGS + wg) * ABLK_USH)
                         + tid * 16;
        unsigned ldsa = (unsigned)(uintptr_t)(smem + LDS_HDR_BYTES) + (unsigned)tid * 16u;
        #pragma unroll 1
        for (int i = 0; i < ABLK_ITERS; ++i) {
            asm volatile("global_load_async_to_lds_b128 %0, %1, off"
                         :: "v"(ldsa), "v"(gsrc) : "memory");
            ldsa += 128 * 16;
            gsrc += 128 * 16;
        }
        asm volatile("s_wait_asynccnt 0" ::: "memory");
    }
    for (int k = tid; k < H2; k += 128) hbuf[k] = 0.f;   // h0 = 0
    __syncthreads();

    float cstate = 0.f;                                  // c for cell (tid<32)
    const float* xgd = xg + (size_t)dir * T_LEN * G4;
    unsigned int* myctr = ctr + dir;
    const unsigned short* wr = wl + tid * WSTRIDE;

    for (int s = 0; s < T_LEN; ++s) {
        // gate[row] = xg[s][row] + W_hh[row] . h_{s-1}
        float acc = xgd[(size_t)s * G4 + row];
        if (s + 1 < T_LEN)                               // hide next step's load
            __builtin_prefetch(&xgd[(size_t)(s + 1) * G4 + row], 0, 3);
        #pragma unroll 4
        for (int k = 0; k < H2; k += 2) {
            const unsigned int p = *(const unsigned int*)(wr + k);    // 2 bf16
            const float2 h2 = *(const float2*)(hbuf + k);
            acc = fmaf(__uint_as_float(p << 16),         h2.x, acc);
            acc = fmaf(__uint_as_float(p & 0xFFFF0000u), h2.y, acc);
        }
        gates[tid] = acc;
        __syncthreads();

        if (tid < 32) {                                  // cell update, wave 0
            const float gi = sigmoidf(gates[cl]);
            const float gf = sigmoidf(gates[32 + cl]);
            const float gg = tanhf(gates[64 + cl]);
            const float go = sigmoidf(gates[96 + cl]);
            cstate = fmaf(gf, cstate, gi * gg);
            const float h = go * tanhf(cstate);
            const int cell = cellBase + cl;
            h_pub[((s & 1) * 2 + dir) * H2 + cell] = h;
            const int tOrig = dir ? (T_LEN - 1 - s) : s; // h_b is time-reversed
            hcat[(size_t)tOrig * HD + dir * H2 + cell] = f2bf(h);
        }
        __threadfence();                                 // make h_pub agent-visible
        __syncthreads();

        if (tid == 0) {                                  // device barrier (per dir)
            __hip_atomic_fetch_add(myctr, 1u, __ATOMIC_RELEASE, __HIP_MEMORY_SCOPE_AGENT);
            const unsigned int tgt = (unsigned int)REC_WGS * (unsigned int)(s + 1);
            while (__hip_atomic_load(myctr, __ATOMIC_ACQUIRE, __HIP_MEMORY_SCOPE_AGENT) < tgt)
                __builtin_amdgcn_s_sleep(2);
        }
        __syncthreads();
        for (int k = tid; k < H2; k += 128)              // pull full h for next step
            hbuf[k] = h_pub[((s & 1) * 2 + dir) * H2 + k];
        __syncthreads();
    }
}

// ---------------------------------------------------------------------------
// 8) Output projection: feats[t][n] = hcat[t] . W2[n] + b[n]  (WMMA bf16)
__global__ void __launch_bounds__(128)
k_gemm_proj(const unsigned short* __restrict__ hcat, const unsigned short* __restrict__ w2s,
            const float* __restrict__ b2, float* __restrict__ feats) {
    const int lane = threadIdx.x & 31;
    const int wv   = threadIdx.x >> 5;
    const int l16  = lane & 15;
    const int hi   = lane >> 4;
    const int mBase = (blockIdx.x * 4 + wv) * 16;
    const int nBase = blockIdx.y * 16;
    const int n = nBase + l16;
    const float bias = (n < KTAG) ? b2[n] : 0.f;

    v8f c = {};
    const unsigned short* aRow = hcat + (size_t)(mBase + l16) * HD;
    const unsigned short* bRow = w2s  + (size_t)n * HD;

    for (int k0 = 0; k0 < HD; k0 += 32) {
        FragBF a, b;
        const unsigned short* ap = aRow + k0 + hi * 8;
        a.q[0] = *(const u32x4*)(ap);
        a.q[1] = *(const u32x4*)(ap + 16);
        const unsigned short* bp = bRow + k0 + hi * 8;
        b.q[0] = *(const u32x4*)(bp);
        b.q[1] = *(const u32x4*)(bp + 16);
        c = __builtin_amdgcn_wmma_f32_16x16x32_bf16(false, a.v, false, b.v,
                                                    (short)0, c, false, false);
    }
    const int mOff = mBase + hi * 8;
    #pragma unroll
    for (int r = 0; r < 8; ++r)
        feats[(size_t)(mOff + r) * KPAD + n] = c[r] + bias;
}

// ---------------------------------------------------------------------------
// 9) CRF forward scan (log partition), single block.
__global__ void k_crf(const float* __restrict__ feats, const float* __restrict__ trans,
                      float* __restrict__ out) {
    __shared__ float tr[KTAG * KTAG];
    __shared__ float alpha[2][KTAG];
    const int tid = threadIdx.x;
    for (int i = tid; i < KTAG * KTAG; i += blockDim.x) tr[i] = trans[i];
    if (tid < KTAG) alpha[0][tid] = (tid == START_TAG) ? 0.f : NEGV;
    __syncthreads();

    for (int t = 0; t < T_LEN; ++t) {
        const int cur = t & 1, nxt = cur ^ 1;
        if (tid < KTAG) {
            const float* trj = tr + tid * KTAG;
            float m = -3.4e38f;
            for (int i = 0; i < KTAG; ++i) m = fmaxf(m, alpha[cur][i] + trj[i]);
            float s = 0.f;
            for (int i = 0; i < KTAG; ++i) s += expf(alpha[cur][i] + trj[i] - m);
            alpha[nxt][tid] = m + logf(s) + feats[(size_t)t * KPAD + tid];
        }
        __syncthreads();
    }
    if (tid == 0) {
        const int cur = T_LEN & 1;
        const float* trS = tr + STOP_TAG * KTAG;
        float m = -3.4e38f;
        for (int i = 0; i < KTAG; ++i) m = fmaxf(m, alpha[cur][i] + trS[i]);
        float s = 0.f;
        for (int i = 0; i < KTAG; ++i) s += expf(alpha[cur][i] + trS[i] - m);
        out[0] = m + logf(s);
    }
}

// ---------------------------------------------------------------------------
extern "C" void kernel_launch(void* const* d_in, const int* in_sizes, int n_in,
                              void* d_out, int out_size, void* d_ws, size_t ws_size,
                              hipStream_t stream) {
    (void)in_sizes; (void)n_in; (void)out_size; (void)ws_size;
    const int*   sent  = (const int*)  d_in[0];
    const float* emb   = (const float*)d_in[1];
    const float* wih_f = (const float*)d_in[2];
    const float* whh_f = (const float*)d_in[3];
    const float* bih_f = (const float*)d_in[4];
    const float* bhh_f = (const float*)d_in[5];
    const float* wih_b = (const float*)d_in[6];
    const float* whh_b = (const float*)d_in[7];
    const float* bih_b = (const float*)d_in[8];
    const float* bhh_b = (const float*)d_in[9];
    const float* w2    = (const float*)d_in[10];
    const float* b2    = (const float*)d_in[11];
    const float* trans = (const float*)d_in[12];
    float* out = (float*)d_out;

    // workspace layout (256B aligned)
    auto align256 = [](size_t x) { return (x + 255) & ~(size_t)255; };
    char* ws = (char*)d_ws;
    size_t off = 0;
    unsigned int* ctr = (unsigned int*)(ws + off); off = align256(off + 8 * 4);
    unsigned short* xbf  = (unsigned short*)(ws + off); off = align256(off + (size_t)T_LEN * EPAD * 2);
    unsigned short* wihS = (unsigned short*)(ws + off); off = align256(off + (size_t)2 * G4 * EPAD * 2);
    float* bsum          = (float*)(ws + off);          off = align256(off + (size_t)2 * G4 * 4);
    unsigned short* w2S  = (unsigned short*)(ws + off); off = align256(off + (size_t)KPAD * HD * 2);
    unsigned short* whhS = (unsigned short*)(ws + off); off = align256(off + (size_t)2 * REC_WGS * ABLK_USH * 2);
    float* h_pub         = (float*)(ws + off);          off = align256(off + (size_t)2 * 2 * H2 * 4);
    float* feats         = (float*)(ws + off);          off = align256(off + (size_t)T_LEN * KPAD * 4);
    unsigned short* hcat = (unsigned short*)(ws + off); off = align256(off + (size_t)T_LEN * HD * 2);
    float* xg            = (float*)(ws + off);          off = align256(off + (size_t)2 * T_LEN * G4 * 4);

    const int smemRec = LDS_HDR_BYTES + ABLK_BYTES;     // 152320 B (< 320KB/WGP)
    (void)hipFuncSetAttribute((const void*)k_lstm_rec,
                              hipFuncAttributeMaxDynamicSharedMemorySize, smemRec);

    k_init<<<1, 64, 0, stream>>>(ctr);
    k_embed<<<T_LEN, 128, 0, stream>>>(sent, emb, xbf);
    k_stage_wih<<<dim3(G4, 2), 128, 0, stream>>>(wih_f, wih_b, wihS);
    k_stage_bias<<<(G4 + 127) / 128, 128, 0, stream>>>(bih_f, bhh_f, bih_b, bhh_b, bsum);
    k_stage_w2<<<KPAD, 128, 0, stream>>>(w2, w2S);
    k_stage_whh<<<dim3(REC_WGS, 2), 256, 0, stream>>>(whh_f, whh_b, whhS);
    k_gemm_in<<<dim3(T_LEN / 16, G4 / 64, 2), 128, 0, stream>>>(xbf, wihS, bsum, xg);
    k_lstm_rec<<<dim3(REC_WGS, 2), 128, smemRec, stream>>>(whhS, xg, h_pub, hcat, ctr);
    k_gemm_proj<<<dim3(T_LEN / 64, KPAD / 16), 128, 0, stream>>>(hcat, w2S, b2, feats);
    k_crf<<<1, 64, 0, stream>>>(feats, trans, out);
}
